// ConvBNN_58213986730443
// MI455X (gfx1250) — 8-bit matrix operand order repaired and checked by reading, compiled, not run on hardware
//
#include <hip/hip_runtime.h>
#include <math.h>


typedef __attribute__((ext_vector_type(8))) int v8i;

__device__ __forceinline__ int fsignf(float v) { return (v > 0.f) - (v < 0.f); }


__global__ void pack_w1(const float* __restrict__ w1, signed char* __restrict__ ws1) {
  int i = blockIdx.x * blockDim.x + threadIdx.x;
  if (i < 800) ws1[i] = (signed char)fsignf(w1[i]);
}

__global__ void pack_w2(const float* __restrict__ w2, int* __restrict__ Bpack2) {
  int idx = blockIdx.x * blockDim.x + threadIdx.x;
  int vg = idx & 7, lane = (idx >> 3) & 31, kblk = (idx >> 8) & 7, ntile = idx >> 11;
  int n = ntile * 16 + (lane & 15), h = lane >> 4;
  int word = 0;
#pragma unroll
  for (int by = 0; by < 4; ++by) {
    int klocal = ((vg >> 2) << 5) + (h << 4) + ((vg & 3) << 2) + by;
    int k = kblk * 64 + klocal;
    int c = k >> 4, t = k & 15;
    float w = w2[((n * 32 + c) << 4) + t];
    word |= (fsignf(w) & 255) << (8 * by);
  }
  Bpack2[idx] = word;
}

__global__ void pack_w3(const float* __restrict__ w3, int* __restrict__ Bpack3) {
  int idx = blockIdx.x * blockDim.x + threadIdx.x;
  int vg = idx & 7, lane = (idx >> 3) & 31, kblk = idx >> 8;
  int n = lane & 15, h = lane >> 4;
  int word = 0;
  if (n < 10) {
#pragma unroll
    for (int by = 0; by < 4; ++by) {
      int klocal = ((vg >> 2) << 5) + (h << 4) + ((vg & 3) << 2) + by;
      int k = kblk * 64 + klocal;
      word |= (fsignf(w3[n * 4096 + k]) & 255) << (8 * by);
    }
  }
  Bpack3[idx] = word;
}

__device__ __forceinline__ void conv1_load(const float* __restrict__ x, int b,
                                           int oy, int ox, int sx[25]) {
#pragma unroll
  for (int dy = 0; dy < 5; ++dy) {
#pragma unroll
    for (int dx = 0; dx < 5; ++dx) {
      int iy = 2 * oy - 2 + dy, ix = 2 * ox - 2 + dx;
      float v = 0.f;
      if (iy >= 0 && iy < 28 && ix >= 0 && ix < 28) v = x[b * 784 + iy * 28 + ix];
      sx[dy * 5 + dx] = fsignf(v);
    }
  }
}

__global__ void __launch_bounds__(256) conv1_stats(const float* __restrict__ x,
                                                   const signed char* __restrict__ ws1,
                                                   int* __restrict__ part1) {
  __shared__ int ssum[32], ssq[32];
  if (threadIdx.x < 32) { ssum[threadIdx.x] = 0; ssq[threadIdx.x] = 0; }
  __syncthreads();
  int tid = blockIdx.x * 256 + threadIdx.x;
  int b = tid / 196, pix = tid % 196;
  int oy = pix / 14, ox = pix % 14;
  int sx[25];
  conv1_load(x, b, oy, ox, sx);
  for (int ch = 0; ch < 32; ++ch) {
    int s = 0;
#pragma unroll
    for (int i = 0; i < 25; ++i) s += sx[i] * (int)ws1[ch * 25 + i];
    atomicAdd(&ssum[ch], s);
    atomicAdd(&ssq[ch], s * s);
  }
  __syncthreads();
  if (threadIdx.x < 32) {
    part1[blockIdx.x * 64 + threadIdx.x * 2 + 0] = ssum[threadIdx.x];
    part1[blockIdx.x * 64 + threadIdx.x * 2 + 1] = ssq[threadIdx.x];
  }
}

__global__ void __launch_bounds__(256) binarize1(const float* __restrict__ x,
                                                 const signed char* __restrict__ ws1,
                                                 const float* __restrict__ ac1,
                                                 signed char* __restrict__ b1) {
  int tid = blockIdx.x * 256 + threadIdx.x;
  int b = tid / 196, pix = tid % 196;
  int oy = pix / 14, ox = pix % 14;
  int sx[25];
  conv1_load(x, b, oy, ox, sx);
  for (int ch = 0; ch < 32; ++ch) {
    int s = 0;
#pragma unroll
    for (int i = 0; i < 25; ++i) s += sx[i] * (int)ws1[ch * 25 + i];
    float v = ac1[2 * ch] * (float)s + ac1[2 * ch + 1];
    b1[(b * 32 + ch) * 196 + pix] = (signed char)fsignf(v);
  }
}

__global__ void __launch_bounds__(128) conv2_gemm(const signed char* __restrict__ b1,
                                                  const int* __restrict__ Bpack2,
                                                  short* __restrict__ h2,
                                                  int* __restrict__ part2) {
  __shared__ signed char tile[32 * 6 * 20];
  __shared__ __align__(32) int afrag[8 * 32 * 8];
  __shared__ int ssum[64], ssq[64];
  const int Mt0 = blockIdx.x * 16;
  const int b = Mt0 >> 6;
  const int p0 = Mt0 & 63;
  const int oy0 = p0 >> 3;
  const int iy0 = 2 * oy0 - 2;

  for (int e = threadIdx.x; e < 32 * 6 * 20; e += 128) {
    int c = e / 120, rem = e % 120, r = rem / 20, col = rem % 20;
    int iy = iy0 + r, ix = col - 2;
    signed char v = 0;
    if (iy >= 0 && iy < 14 && ix >= 0 && ix < 14)
      v = b1[(b * 32 + c) * 196 + iy * 14 + ix];
    tile[e] = v;
  }
  if (threadIdx.x < 64) { ssum[threadIdx.x] = 0; ssq[threadIdx.x] = 0; }
  __syncthreads();

  {
    int base = threadIdx.x * 16;
#pragma unroll
    for (int j = 0; j < 16; ++j) {
      int e = base + j;
      int kblk = e >> 8, lane2 = (e >> 3) & 31, vg = e & 7;
      int mrow = lane2 & 15, h2i = lane2 >> 4;
      int dy = mrow >> 3, ox = mrow & 7;
      int klocal = ((vg >> 2) << 5) + (h2i << 4) + ((vg & 3) << 2);
      int k = kblk * 64 + klocal;
      int c = k >> 4, ky = (k >> 2) & 3;
      const signed char* p = &tile[c * 120 + (2 * dy + ky) * 20 + 2 * ox];
      unsigned lo = *(const unsigned short*)(p);
      unsigned hi = *(const unsigned short*)(p + 2);
      afrag[e] = (int)(lo | (hi << 16));
    }
  }
  __syncthreads();

  const int lane = threadIdx.x & 31;
  const int wave = threadIdx.x >> 5;

  v8i acc = {0, 0, 0, 0, 0, 0, 0, 0};
#pragma unroll
  for (int kblk = 0; kblk < 8; ++kblk) {
    v8i a = *(const v8i*)&afrag[(kblk * 32 + lane) * 8];
    v8i bf = *(const v8i*)(Bpack2 + ((wave * 8 + kblk) * 32 + lane) * 8);
    acc = __builtin_amdgcn_wmma_i32_16x16x64_iu8(true, a, true, bf, acc, false, false);
  }

  const int n = wave * 16 + (lane & 15);
  const int hd = lane >> 4;
  int ls = 0, lq = 0;
#pragma unroll
  for (int r = 0; r < 8; ++r) {
    int v = acc[r];
    int p = p0 + r + 8 * hd;
    h2[(b * 64 + n) * 64 + p] = (short)v;
    ls += v;
    lq += v * v;
  }
  atomicAdd(&ssum[n], ls);
  atomicAdd(&ssq[n], lq);
  __syncthreads();
  if (threadIdx.x < 64) {
    part2[blockIdx.x * 128 + threadIdx.x * 2 + 0] = ssum[threadIdx.x];
    part2[blockIdx.x * 128 + threadIdx.x * 2 + 1] = ssq[threadIdx.x];
  }
}

__global__ void __launch_bounds__(256) binarize2(const short* __restrict__ h2,
                                                 const float* __restrict__ ac2,
                                                 signed char* __restrict__ b2) {
  int i = blockIdx.x * 256 + threadIdx.x;
  int ch = (i >> 6) & 63;
  float v = ac2[2 * ch] * (float)h2[i] + ac2[2 * ch + 1];
  b2[i] = (signed char)fsignf(v);
}

__global__ void __launch_bounds__(32) fc3_gemm(const signed char* __restrict__ b2,
                                               const int* __restrict__ Bpack3,
                                               int* __restrict__ yint,
                                               long long* __restrict__ part3) {
  __shared__ long long shs[16], shq[16];
  const int Mt0 = blockIdx.x * 16;
  const int lane = threadIdx.x;
  const int h = lane >> 4;
  const signed char* arow = b2 + (size_t)(Mt0 + (lane & 15)) * 4096;

  v8i acc = {0, 0, 0, 0, 0, 0, 0, 0};
  for (int kblk = 0; kblk < 64; ++kblk) {
    __builtin_prefetch(arow + (kblk + 1) * 64, 0, 1);
    v8i a;
#pragma unroll
    for (int vg = 0; vg < 8; ++vg) {
      int klocal = ((vg >> 2) << 5) + (h << 4) + ((vg & 3) << 2);
      a[vg] = *(const int*)(arow + kblk * 64 + klocal);
    }
    v8i bf = *(const v8i*)(Bpack3 + (kblk * 32 + lane) * 8);
    acc = __builtin_amdgcn_wmma_i32_16x16x64_iu8(true, a, true, bf, acc, false, false);
  }

  const int n = lane & 15;
  long long ls = 0, lq = 0;
#pragma unroll
  for (int r = 0; r < 8; ++r) {
    int v = acc[r];
    yint[(Mt0 + r + 8 * h) * 16 + n] = v;
    ls += v;
    lq += (long long)v * v;
  }
  if (h == 0) { shs[n] = ls; shq[n] = lq; }
  __syncthreads();
  if (h == 1) { shs[n] += ls; shq[n] += lq; }
  __syncthreads();
  if (h == 0) {
    part3[blockIdx.x * 32 + n * 2 + 0] = shs[n];
    part3[blockIdx.x * 32 + n * 2 + 1] = shq[n];
  }
}

__global__ void reduce_i32_i64(const int* __restrict__ src, int nblocks, int stride,
                               long long* __restrict__ dst) {
  __shared__ long long sh[256];
  long long s = 0;
  for (int j = threadIdx.x; j < nblocks; j += 256) s += (long long)src[j * stride + blockIdx.x];
  sh[threadIdx.x] = s;
  __syncthreads();
  for (int o = 128; o > 0; o >>= 1) {
    if ((int)threadIdx.x < o) sh[threadIdx.x] += sh[threadIdx.x + o];
    __syncthreads();
  }
  if (threadIdx.x == 0) dst[blockIdx.x] = sh[0];
}

__global__ void reduce_i64(const long long* __restrict__ src, int nblocks, int stride,
                           long long* __restrict__ dst) {
  __shared__ long long sh[256];
  long long s = 0;
  for (int j = threadIdx.x; j < nblocks; j += 256) s += src[j * stride + blockIdx.x];
  sh[threadIdx.x] = s;
  __syncthreads();
  for (int o = 128; o > 0; o >>= 1) {
    if ((int)threadIdx.x < o) sh[threadIdx.x] += sh[threadIdx.x + o];
    __syncthreads();
  }
  if (threadIdx.x == 0) dst[blockIdx.x] = sh[0];
}

__global__ void bn_params(const long long* __restrict__ red, const float* __restrict__ g,
                          const float* __restrict__ beta, float count,
                          float* __restrict__ ac, int C) {
  int c = threadIdx.x;
  if (c >= C) return;
  double mean = (double)red[2 * c] / (double)count;
  double var = (double)red[2 * c + 1] / (double)count - mean * mean;
  float a = g[c] * (float)(1.0 / sqrt(var + 1e-5));
  ac[2 * c] = a;
  ac[2 * c + 1] = beta[c] - (float)mean * a;
}

__global__ void final_out(const int* __restrict__ yint, const float* __restrict__ ac3,
                          float* __restrict__ out) {
  int i = blockIdx.x * 256 + threadIdx.x;
  int b = i / 10, j = i % 10;
  out[i] = ac3[2 * j] * (float)yint[b * 16 + j] + ac3[2 * j + 1];
}

extern "C" void kernel_launch(void* const* d_in, const int* in_sizes, int n_in,
                              void* d_out, int out_size, void* d_ws, size_t ws_size,
                              hipStream_t stream) {
  const float* x  = (const float*)d_in[0];
  const float* w1 = (const float*)d_in[1];
  const float* g1 = (const float*)d_in[2];
  const float* be1 = (const float*)d_in[3];
  const float* w2 = (const float*)d_in[4];
  const float* g2 = (const float*)d_in[5];
  const float* be2 = (const float*)d_in[6];
  const float* w3 = (const float*)d_in[7];
  const float* g3 = (const float*)d_in[8];
  const float* be3 = (const float*)d_in[9];
  float* out = (float*)d_out;
  char* ws = (char*)d_ws;

  signed char* b1    = (signed char*)(ws + 0);
  short*       h2    = (short*)(ws + 51380224);
  signed char* b2    = (signed char*)(ws + 118489088);
  int*         yint  = (int*)(ws + 152043520);
  int*         Bp2   = (int*)(ws + 152567808);
  int*         Bp3   = (int*)(ws + 152600576);
  signed char* ws1   = (signed char*)(ws + 152666112);
  int*         part1 = (int*)(ws + 152667136);
  int*         part2 = (int*)(ws + 154272768);
  long long*   part3 = (long long*)(ws + 171049984);
  long long*   red1  = (long long*)(ws + 171181056);
  long long*   red2  = (long long*)(ws + 171181568);
  long long*   red3  = (long long*)(ws + 171182592);
  float*       ac1   = (float*)(ws + 171182848);
  float*       ac2   = (float*)(ws + 171183104);
  float*       ac3   = (float*)(ws + 171183616);

  pack_w1<<<4, 256, 0, stream>>>(w1, ws1);
  pack_w2<<<32, 256, 0, stream>>>(w2, Bp2);
  pack_w3<<<64, 256, 0, stream>>>(w3, Bp3);

  conv1_stats<<<6272, 256, 0, stream>>>(x, ws1, part1);
  reduce_i32_i64<<<64, 256, 0, stream>>>(part1, 6272, 64, red1);
  bn_params<<<1, 64, 0, stream>>>(red1, g1, be1, 8192.f * 196.f, ac1, 32);
  binarize1<<<6272, 256, 0, stream>>>(x, ws1, ac1, b1);

  conv2_gemm<<<32768, 128, 0, stream>>>(b1, Bp2, h2, part2);
  reduce_i32_i64<<<128, 256, 0, stream>>>(part2, 32768, 128, red2);
  bn_params<<<1, 64, 0, stream>>>(red2, g2, be2, 8192.f * 64.f, ac2, 64);
  binarize2<<<131072, 256, 0, stream>>>(h2, ac2, b2);

  fc3_gemm<<<512, 32, 0, stream>>>(b2, Bp3, yint, part3);
  reduce_i64<<<32, 256, 0, stream>>>(part3, 512, 32, red3);
  bn_params<<<1, 16, 0, stream>>>(red3, g3, be3, 8192.f, ac3, 10);
  final_out<<<320, 256, 0, stream>>>(yint, ac3, out);
}
